// RegEncodeBlock_50895362458114
// MI455X (gfx1250) — matrix operand order repaired and checked by reading, not compiled, not run on hardware
//
#include <hip/hip_runtime.h>
#include <hip/hip_bf16.h>

#define D_    32
#define H_    128
#define W_    160
#define CIN_  16
#define COUT_ 32
#define HW_   (H_*W_)
#define DHW_  (D_*H_*W_)
#define EPS_  1e-5f

typedef __attribute__((ext_vector_type(16))) __bf16 v16bf;
typedef __attribute__((ext_vector_type(8)))  __bf16 v8bf;
typedef __attribute__((ext_vector_type(8)))  float  v8f;

// ---------------------------------------------------------------------------
// Zero a small stats region.
// ---------------------------------------------------------------------------
__global__ void zero_kernel(float* __restrict__ p, int n) {
    int i = blockIdx.x * 256 + threadIdx.x;
    if (i < n) p[i] = 0.f;
}

// ---------------------------------------------------------------------------
// Deformable conv2d per depth slice: offset conv (18ch) -> clip -> bilinear
// gather (9 taps x 16ch) -> 32x144 mat-vec. Writes raw out1 [32][D][H][W] and
// accumulates per-(d,c) sum/sumsq for BN2d via wave shuffle-reduce + atomics.
// ---------------------------------------------------------------------------
__global__ __launch_bounds__(256)
void deform_kernel(const float* __restrict__ x, const float* __restrict__ f,
                   const float* __restrict__ w_off, const float* __restrict__ w_reg,
                   float* __restrict__ out1, float* __restrict__ sum1,
                   float* __restrict__ sumsq1)
{
    __shared__ float woff_s[18 * 16 * 9];   // [o18][ci16][9]
    __shared__ float wreg_s[32 * 16 * 9];   // [o32][ci16][t9]
    __shared__ float sst[64];

    const int tid = threadIdx.x;
    for (int i = tid; i < 18 * 16 * 9; i += 256) woff_s[i] = w_off[i];
    for (int i = tid; i < 32 * 16 * 9; i += 256) wreg_s[i] = w_reg[i];
    if (tid < 64) sst[tid] = 0.f;
    __syncthreads();

    const int d = blockIdx.y;
    const int p = blockIdx.x * 256 + tid;       // pixel within slice
    const int h = p / W_, w = p % W_;

    // ---- offset conv2d (pad=1), 18 output channels ----
    float off[18];
#pragma unroll
    for (int o = 0; o < 18; ++o) off[o] = 0.f;
#pragma unroll
    for (int dh = 0; dh < 3; ++dh) {
        const int hy = h + dh - 1;
        if (hy < 0 || hy >= H_) continue;
#pragma unroll
        for (int dw = 0; dw < 3; ++dw) {
            const int wx = w + dw - 1;
            if (wx < 0 || wx >= W_) continue;
            const int ko = dh * 3 + dw;
            for (int ci = 0; ci < CIN_; ++ci) {
                const float fv = f[((ci * D_ + d) * H_ + hy) * W_ + wx];
#pragma unroll
                for (int o = 0; o < 18; ++o)
                    off[o] += fv * woff_s[(o * CIN_ + ci) * 9 + ko];
            }
        }
    }
#pragma unroll
    for (int o = 0; o < 18; ++o) off[o] = fminf(fmaxf(off[o], -1.f), 1.f);

    // ---- bilinear sample 9 taps, accumulate 32 outputs ----
    float acc[32];
#pragma unroll
    for (int o = 0; o < 32; ++o) acc[o] = 0.f;

    for (int t = 0; t < 9; ++t) {
        const float py = (float)(h - 1 + t / 3) + off[2 * t];
        const float px = (float)(w - 1 + t % 3) + off[2 * t + 1];
        const float y0f = floorf(py), x0f = floorf(px);
        const float wy1 = py - y0f, wx1 = px - x0f;
        const int y0 = (int)y0f, x0 = (int)x0f;
        const int y1 = y0 + 1, x1 = x0 + 1;
        const float my0 = (y0 >= 0 && y0 < H_) ? 1.f : 0.f;
        const float my1 = (y1 >= 0 && y1 < H_) ? 1.f : 0.f;
        const float mx0 = (x0 >= 0 && x0 < W_) ? 1.f : 0.f;
        const float mx1 = (x1 >= 0 && x1 < W_) ? 1.f : 0.f;
        const int y0c = min(max(y0, 0), H_ - 1), y1c = min(max(y1, 0), H_ - 1);
        const int x0c = min(max(x0, 0), W_ - 1), x1c = min(max(x1, 0), W_ - 1);
        const float w00 = (1.f - wy1) * (1.f - wx1) * my0 * mx0;
        const float w01 = (1.f - wy1) * wx1 * my0 * mx1;
        const float w10 = wy1 * (1.f - wx1) * my1 * mx0;
        const float w11 = wy1 * wx1 * my1 * mx1;
        for (int ci = 0; ci < CIN_; ++ci) {
            const float* xb = x + ((size_t)(ci * D_ + d) * H_) * W_;
            const float s = w00 * xb[y0c * W_ + x0c] + w01 * xb[y0c * W_ + x1c]
                          + w10 * xb[y1c * W_ + x0c] + w11 * xb[y1c * W_ + x1c];
#pragma unroll
            for (int o = 0; o < 32; ++o)
                acc[o] += s * wreg_s[(o * CIN_ + ci) * 9 + t];
        }
    }

    // ---- store raw + per-(d,c) stats ----
    const int lane = tid & 31;
#pragma unroll
    for (int o = 0; o < 32; ++o) {
        const float v = acc[o];
        out1[(size_t)(o * D_ + d) * HW_ + p] = v;
        float s = v, q = v * v;
#pragma unroll
        for (int m = 16; m; m >>= 1) { s += __shfl_xor(s, m); q += __shfl_xor(q, m); }
        if (lane == 0) { atomicAdd(&sst[o], s); atomicAdd(&sst[32 + o], q); }
    }
    __syncthreads();
    if (tid < 32) {
        atomicAdd(&sum1[d * 32 + tid], sst[tid]);
        atomicAdd(&sumsq1[d * 32 + tid], sst[32 + tid]);
    }
}

// ---------------------------------------------------------------------------
// BN coefficient kernels.
// ---------------------------------------------------------------------------
__global__ void stats1_kernel(const float* __restrict__ sum1, const float* __restrict__ sumsq1,
                              const float* __restrict__ g, const float* __restrict__ b,
                              float* __restrict__ scale1, float* __restrict__ bias1)
{
    int i = blockIdx.x * 256 + threadIdx.x;
    if (i >= D_ * 32) return;
    const int c = i & 31;
    const float inv = 1.f / (float)HW_;
    const float mean = sum1[i] * inv;
    const float var  = sumsq1[i] * inv - mean * mean;
    const float s = g[c] * rsqrtf(var + EPS_);
    scale1[i] = s;
    bias1[i]  = b[c] - mean * s;
}

__global__ void stats2_kernel(const float* __restrict__ sum2, const float* __restrict__ sumsq2,
                              const float* __restrict__ g2, const float* __restrict__ b2,
                              const float* __restrict__ sumd, const float* __restrict__ sumsqd,
                              const float* __restrict__ gd, const float* __restrict__ bd,
                              float* __restrict__ scale2, float* __restrict__ bias2,
                              float* __restrict__ scaled, float* __restrict__ biasd)
{
    const int i = threadIdx.x;
    const float inv = 1.f / (float)DHW_;
    if (i < 32) {
        const float mean = sum2[i] * inv, var = sumsq2[i] * inv - mean * mean;
        const float s = g2[i] * rsqrtf(var + EPS_);
        scale2[i] = s; bias2[i] = b2[i] - mean * s;
    } else if (i < 64) {
        const int c = i - 32;
        const float mean = sumd[c] * inv, var = sumsqd[c] * inv - mean * mean;
        const float s = gd[c] * rsqrtf(var + EPS_);
        scaled[c] = s; biasd[c] = bd[c] - mean * s;
    }
}

// ---------------------------------------------------------------------------
// Implicit-GEMM 3x3x3 conv3d via v_wmma_f32_16x16x32_bf16 (wave32).
// One workgroup per output (d,h) row; 10 waves, wave i owns w-tile [16i,16i+16).
// M = 16 w positions, N = 16 couts (2 n-tiles), K = cin per tap (27 taps).
// Optional fused BN+ReLU on the input (per-(dz,cin) scale/bias).
// Writes raw conv output and accumulates per-cout sum/sumsq.
// ---------------------------------------------------------------------------
template <int CIN>
__global__ __launch_bounds__(320)
void conv3d_wmma_kernel(const float* __restrict__ in, const float* __restrict__ wgt,
                        const float* __restrict__ sc, const float* __restrict__ bi,
                        float* __restrict__ out, float* __restrict__ sum,
                        float* __restrict__ sumsq)
{
    extern __shared__ __align__(64) char smem[];
    __bf16* in_s = (__bf16*)smem;                              // [9][162][CIN]
    __bf16* w_s  = (__bf16*)(smem + (size_t)9 * 162 * CIN * 2); // [27][32][32], k>=CIN zero
    float*  sst  = (float*)(smem + (size_t)9 * 162 * CIN * 2 + 27 * 32 * 32 * 2); // [64]

    const int tid = threadIdx.x;
    const int h = blockIdx.x, d = blockIdx.y;

    if (tid < 64) sst[tid] = 0.f;

    // stage weights -> LDS [tap][n][k] (bf16, zero-padded k)
    for (int i = tid; i < 27 * 32 * 32; i += 320) {
        const int k = i & 31, n = (i >> 5) & 31, t = i >> 10;
        float v = 0.f;
        if (k < CIN) {
            const int kd = t / 9, kh = (t / 3) % 3, kw = t % 3;
            v = wgt[(((n * CIN + k) * 3 + kd) * 3 + kh) * 3 + kw];
        }
        w_s[i] = (__bf16)v;
    }

    // stage 9 input rows (3 d x 3 h), width W+2, channel-innermost, with
    // optional fused BN(scale per (dz,cin)) + ReLU, converted to bf16.
    for (int i = tid; i < 9 * 162 * CIN; i += 320) {
        const int k  = i % CIN;
        const int wx = (i / CIN) % 162;
        const int r  = i / (CIN * 162);
        const int dz = d + r / 3 - 1, hy = h + r % 3 - 1, wxx = wx - 1;
        float v = 0.f;
        if (dz >= 0 && dz < D_ && hy >= 0 && hy < H_ && wxx >= 0 && wxx < W_) {
            v = in[((size_t)(k * D_ + dz) * H_ + hy) * W_ + wxx];
            if (sc) v = fmaxf(v * sc[dz * 32 + k] + bi[dz * 32 + k], 0.f);
        }
        in_s[i] = (__bf16)v;
    }
    __syncthreads();

    const int wv = tid >> 5, lane = tid & 31;
    const int m = lane & 15, half = lane >> 4;   // A: M=lane%16; B/D: N=lane%16
    const int w0 = wv * 16;

    v8f acc0 = {}, acc1 = {};

    for (int t = 0; t < 27; ++t) {
        const int kd = t / 9, kh = (t / 3) % 3, kw = t % 3;
        const int r = kd * 3 + kh;

        // A fragment: 16-bit A 16x32 layout — lanes<16: K {0..7,16..23},
        // lanes>=16: K {8..15,24..31}; K = cin (zero-padded for CIN=16).
        const __bf16* ap = &in_s[((r * 162) + (w0 + m + kw)) * CIN];
        v8bf alo = *(const v8bf*)(ap + half * 8);
        v8bf ahi = {};
        if (CIN == 32) ahi = *(const v8bf*)(ap + 16 + half * 8);
        v16bf a = __builtin_shufflevector(alo, ahi,
                      0, 1, 2, 3, 4, 5, 6, 7, 8, 9, 10, 11, 12, 13, 14, 15);

        // B fragment: lanes<16: K {0..7,16..23}, lanes>=16: K {8..15,24..31},
        // N = lane%16 -> two 16B reads of w_s[t][n][half*8 ..] and [16+half*8 ..].
        v16bf b0 = __builtin_shufflevector(
                      *(const v8bf*)&w_s[((t * 32) + m) * 32 + half * 8],
                      *(const v8bf*)&w_s[((t * 32) + m) * 32 + 16 + half * 8],
                      0, 1, 2, 3, 4, 5, 6, 7, 8, 9, 10, 11, 12, 13, 14, 15);
        acc0 = __builtin_amdgcn_wmma_f32_16x16x32_bf16(
                   false, a, false, b0, (short)0, acc0, false, false);

        v16bf b1 = __builtin_shufflevector(
                      *(const v8bf*)&w_s[((t * 32) + 16 + m) * 32 + half * 8],
                      *(const v8bf*)&w_s[((t * 32) + 16 + m) * 32 + 16 + half * 8],
                      0, 1, 2, 3, 4, 5, 6, 7, 8, 9, 10, 11, 12, 13, 14, 15);
        acc1 = __builtin_amdgcn_wmma_f32_16x16x32_bf16(
                   false, a, false, b1, (short)0, acc1, false, false);
    }

    // D layout: lane -> N=lane%16, VGPR r -> M=r+8*(lane/16): 8 consecutive w.
    {
        const int c0 = m, c1 = 16 + m;
        float* o0 = &out[((size_t)(c0 * D_ + d) * H_ + h) * W_ + w0 + half * 8];
        float* o1 = &out[((size_t)(c1 * D_ + d) * H_ + h) * W_ + w0 + half * 8];
        ((float4*)o0)[0] = make_float4(acc0[0], acc0[1], acc0[2], acc0[3]);
        ((float4*)o0)[1] = make_float4(acc0[4], acc0[5], acc0[6], acc0[7]);
        ((float4*)o1)[0] = make_float4(acc1[0], acc1[1], acc1[2], acc1[3]);
        ((float4*)o1)[1] = make_float4(acc1[4], acc1[5], acc1[6], acc1[7]);

        float s0 = 0.f, q0 = 0.f, s1 = 0.f, q1 = 0.f;
#pragma unroll
        for (int r2 = 0; r2 < 8; ++r2) {
            s0 += acc0[r2]; q0 += acc0[r2] * acc0[r2];
            s1 += acc1[r2]; q1 += acc1[r2] * acc1[r2];
        }
        atomicAdd(&sst[c0], s0); atomicAdd(&sst[32 + c0], q0);
        atomicAdd(&sst[c1], s1); atomicAdd(&sst[32 + c1], q1);
    }
    __syncthreads();
    if (tid < 32) {
        atomicAdd(&sum[tid], sst[tid]);
        atomicAdd(&sumsq[tid], sst[32 + tid]);
    }
}

// ---------------------------------------------------------------------------
// Epilogue: out = relu(BN2(out2_raw) + BNd(id_raw)), in place on d_out.
// ---------------------------------------------------------------------------
__global__ __launch_bounds__(256)
void final_kernel(float* __restrict__ out, const float* __restrict__ idraw,
                  const float* __restrict__ scale2, const float* __restrict__ bias2,
                  const float* __restrict__ scaled, const float* __restrict__ biasd)
{
    const int i = blockIdx.x * 256 + threadIdx.x;   // float4 index
    const int c = i / (DHW_ / 4);
    const float4 o = ((const float4*)out)[i];
    const float4 v = ((const float4*)idraw)[i];
    const float s2 = scale2[c], b2 = bias2[c], sd = scaled[c], bd = biasd[c];
    float4 r;
    r.x = fmaxf(o.x * s2 + b2 + v.x * sd + bd, 0.f);
    r.y = fmaxf(o.y * s2 + b2 + v.y * sd + bd, 0.f);
    r.z = fmaxf(o.z * s2 + b2 + v.z * sd + bd, 0.f);
    r.w = fmaxf(o.w * s2 + b2 + v.w * sd + bd, 0.f);
    ((float4*)out)[i] = r;
}

// ---------------------------------------------------------------------------
extern "C" void kernel_launch(void* const* d_in, const int* in_sizes, int n_in,
                              void* d_out, int out_size, void* d_ws, size_t ws_size,
                              hipStream_t stream)
{
    (void)in_sizes; (void)n_in; (void)out_size; (void)ws_size;

    const float* x     = (const float*)d_in[0];
    const float* f     = (const float*)d_in[1];
    const float* w_off = (const float*)d_in[2];
    const float* w_reg = (const float*)d_in[3];
    const float* g1    = (const float*)d_in[4];
    const float* b1    = (const float*)d_in[5];
    const float* w2    = (const float*)d_in[6];
    const float* g2    = (const float*)d_in[7];
    const float* b2    = (const float*)d_in[8];
    const float* wd    = (const float*)d_in[9];
    const float* gd    = (const float*)d_in[10];
    const float* bd    = (const float*)d_in[11];
    float* out = (float*)d_out;

    float* ws = (float*)d_ws;
    const size_t VOL = (size_t)COUT_ * DHW_;
    float* out1   = ws;                  // raw deformable output [32][D][H][W]
    float* idraw  = out1 + VOL;          // raw downsample output
    float* sum1   = idraw + VOL;         // [D*32] -- start of zeroed stats block
    float* sumsq1 = sum1 + 1024;
    float* sum2   = sumsq1 + 1024;       // [32]
    float* sumsq2 = sum2 + 32;
    float* sumd   = sumsq2 + 32;
    float* sumsqd = sumd + 32;           // zero block = 2*1024 + 4*32 = 2176
    float* scale1 = sumsqd + 32;         // [D*32]
    float* bias1  = scale1 + 1024;
    float* scale2 = bias1 + 1024;        // [32] x4
    float* bias2  = scale2 + 32;
    float* scaled = bias2 + 32;
    float* biasd  = scaled + 32;

    zero_kernel<<<(2176 + 255) / 256, 256, 0, stream>>>(sum1, 2176);

    dim3 gdef(HW_ / 256, D_);
    deform_kernel<<<gdef, 256, 0, stream>>>(x, f, w_off, w_reg, out1, sum1, sumsq1);

    stats1_kernel<<<4, 256, 0, stream>>>(sum1, sumsq1, g1, b1, scale1, bias1);

    dim3 gc(H_, D_);
    const size_t smem32 = (size_t)9 * 162 * 32 * 2 + 27 * 32 * 32 * 2 + 64 * 4;
    const size_t smem16 = (size_t)9 * 162 * 16 * 2 + 27 * 32 * 32 * 2 + 64 * 4;
    // conv3d(w2) on BN1+ReLU(out1); raw result -> d_out, stats -> sum2
    conv3d_wmma_kernel<32><<<gc, 320, smem32, stream>>>(
        out1, w2, scale1, bias1, out, sum2, sumsq2);
    // downsample conv3d(wd) on x; raw result -> idraw, stats -> sumd
    conv3d_wmma_kernel<16><<<gc, 320, smem16, stream>>>(
        x, wd, nullptr, nullptr, idraw, sumd, sumsqd);

    stats2_kernel<<<1, 64, 0, stream>>>(sum2, sumsq2, g2, b2,
                                        sumd, sumsqd, gd, bd,
                                        scale2, bias2, scaled, biasd);

    final_kernel<<<((COUT_ * DHW_) / 4) / 256, 256, 0, stream>>>(
        out, idraw, scale2, bias2, scaled, biasd);
}
